// HierarchicalHopfieldLayer_41214506173037
// MI455X (gfx1250) — matrix operand order repaired and checked by reading, not compiled, not run on hardware
//
#include <hip/hip_runtime.h>
#include <hip/hip_bf16.h>

// ---------------------------------------------------------------------------
// HierarchicalHopfieldLayer on MI455X (gfx1250, wave32, WMMA bf16 16x16x32)
// B=2 S=512 Dh=2048 N=8192 Dm=1024 H=4 D=64 T=256, 3 Hopfield steps
// ---------------------------------------------------------------------------

typedef __attribute__((ext_vector_type(16))) __bf16 v16bf;
typedef __attribute__((ext_vector_type(8)))  __bf16 bf16x8;
typedef __attribute__((ext_vector_type(8)))  float  v8f;

#define HN 8192   // memory slots
#define HT 256    // H*D

__device__ __forceinline__ unsigned short f2h_bits(float x){ _Float16 h=(_Float16)x; return __builtin_bit_cast(unsigned short,h); }
__device__ __forceinline__ float h2f(unsigned short u){ _Float16 h=__builtin_bit_cast(_Float16,u); return (float)h; }
__device__ __forceinline__ unsigned short f2b_bits(float x){ __bf16 h=(__bf16)x; return __builtin_bit_cast(unsigned short,h); }

// ------------------- f32 [R][C] -> bf16 [C][R] (weight prep) ---------------
__global__ void k_transpose_bf16(const float* __restrict__ in,
                                 unsigned short* __restrict__ out, int R, int C){
  int idx = blockIdx.x * 256 + threadIdx.x;
  if (idx >= R * C) return;
  int r = idx / C, c = idx - r * C;
  out[(size_t)c * R + r] = f2b_bits(in[idx]);
}

// ------------------- row LayerNorm -> bf16 ---------------------------------
__global__ void k_ln_bf16(const float* __restrict__ x, const float* __restrict__ g,
                          const float* __restrict__ b, unsigned short* __restrict__ out, int L){
  __shared__ float s1[256], s2[256];
  int row = blockIdx.x, tid = threadIdx.x;
  const float* xr = x + (size_t)row * L;
  float a = 0.f, a2 = 0.f;
  for (int i = tid; i < L; i += 256){ float v = xr[i]; a += v; a2 += v * v; }
  s1[tid] = a; s2[tid] = a2; __syncthreads();
  for (int off = 128; off > 0; off >>= 1){
    if (tid < off){ s1[tid] += s1[tid + off]; s2[tid] += s2[tid + off]; }
    __syncthreads();
  }
  float mu = s1[0] / (float)L;
  float var = s2[0] / (float)L - mu * mu;
  float rs = rsqrtf(var + 1e-5f);
  unsigned short* orow = out + (size_t)row * L;
  for (int i = tid; i < L; i += 256) orow[i] = f2b_bits((xr[i] - mu) * rs * g[i] + b[i]);
}

// ------------------- generic WMMA GEMM: C[M][N] = A[M][K] * Bt[N][K]^T -----
// mode 0: C f32 row-major | mode 1: C bf16 row-major | mode 2: C bf16 [N][M]
__global__ void k_gemm_bf16(const unsigned short* __restrict__ A,
                            const unsigned short* __restrict__ Bt,
                            float* __restrict__ Cf, unsigned short* __restrict__ Cb,
                            int M, int N, int K, int mode){
  int lane = threadIdx.x & 31;
  int wave = threadIdx.x >> 5;
  int m0 = blockIdx.x * 16;
  int n0 = blockIdx.y * 128 + wave * 16;
  int ln = lane & 15, lh = lane >> 4;
  const unsigned short* arow = A  + (size_t)(m0 + ln) * K;
  const unsigned short* brow = Bt + (size_t)(n0 + ln) * K;
  v8f acc = {};
  for (int k0 = 0; k0 < K; k0 += 32){
    bf16x8 alo = *reinterpret_cast<const bf16x8*>(arow + k0 + 16 * lh);
    bf16x8 ahi = *reinterpret_cast<const bf16x8*>(arow + k0 + 16 * lh + 8);
    v16bf a;
#pragma unroll
    for (int e = 0; e < 8; e++){ a[e] = alo[e]; a[e + 8] = ahi[e]; }
    v16bf bfr = *reinterpret_cast<const v16bf*>(brow + k0 + 16 * lh);
    acc = __builtin_amdgcn_wmma_f32_16x16x32_bf16(false, a, false, bfr, (short)0, acc, false, false);
  }
#pragma unroll
  for (int r = 0; r < 8; r++){
    int m = m0 + 8 * lh + r, n = n0 + ln;
    if (mode == 0)      Cf[(size_t)m * N + n] = acc[r];
    else if (mode == 1) Cb[(size_t)m * N + n] = f2b_bits(acc[r]);
    else                Cb[(size_t)n * M + m] = f2b_bits(acc[r]);
  }
}

// ------------------- fused 3-step Hopfield with sort-free entmax-1.5 -------
// One block = 16 query rows of one (b,h). Scores (16x8192 f16) live in LDS.
// K stored bf16 [N][T]; V stored transposed bf16 [T][N].
__global__ void k_hopfield(const float* __restrict__ Q, const unsigned short* __restrict__ Kb,
                           const unsigned short* __restrict__ Vt, const float* __restrict__ log_beta,
                           float* __restrict__ conv_f, unsigned short* __restrict__ conv_b){
  extern __shared__ char smem[];
  unsigned short* sc = (unsigned short*)smem;            // 16*8192 f16 -> later bf16 weights
  float* xi    = (float*)(smem + 16 * HN * 2);           // 16*64
  float* red   = xi + 16 * 64;                           // 16*16
  float* maxs  = red + 256;                              // 16
  float* taus  = maxs + 16;                              // 16
  float* cpart = taus + 16;                              // 8*16*16

  int tid = threadIdx.x;
  int lane = tid & 31, wave = tid >> 5;
  int ln = lane & 15, lh = lane >> 4;
  int bid = blockIdx.x;
  int st = bid & 31, hh = (bid >> 5) & 3, bb = bid >> 7;
  int s0 = st * 16;
  int rowbase = bb * 512 + s0;
  float beta = __expf(log_beta[hh]);

  // init xi from Q [BS][256]
  for (int o = tid; o < 16 * 64; o += 256){
    int m = o >> 6, d = o & 63;
    xi[m * 64 + d] = Q[(size_t)(rowbase + m) * HT + hh * 64 + d];
  }
  __syncthreads();

  for (int step = 0; step < 3; step++){
    // ---- GEMM1: scores[16][8192] = xi @ K^T * beta (bf16 WMMA) ----
    v16bf a0, a1;
    {
      const float* xr = xi + ln * 64;
#pragma unroll
      for (int e = 0; e < 8; e++){
        a0[e]     = (__bf16)xr[16 * lh + e];
        a0[e + 8] = (__bf16)xr[8 + 16 * lh + e];
        a1[e]     = (__bf16)xr[32 + 16 * lh + e];
        a1[e + 8] = (__bf16)xr[40 + 16 * lh + e];
      }
    }
    for (int t = wave; t < HN / 16; t += 8){
      int n0 = t * 16;
      const unsigned short* kp = Kb + (size_t)(n0 + ln) * HT + hh * 64;
      v16bf b0 = *reinterpret_cast<const v16bf*>(kp + 16 * lh);
      v16bf b1 = *reinterpret_cast<const v16bf*>(kp + 32 + 16 * lh);
      v8f c = {};
      c = __builtin_amdgcn_wmma_f32_16x16x32_bf16(false, a0, false, b0, (short)0, c, false, false);
      c = __builtin_amdgcn_wmma_f32_16x16x32_bf16(false, a1, false, b1, (short)0, c, false, false);
#pragma unroll
      for (int r = 0; r < 8; r++)
        sc[(8 * lh + r) * HN + n0 + ln] = f2h_bits(c[r] * beta);
    }
    __syncthreads();

    // ---- per-row max (16 threads per row, strided) ----
    int row = tid >> 4, j = tid & 15;
    const unsigned short* srow = sc + row * HN;
    float mx = -3.0e38f;
    for (int i = j; i < HN; i += 16) mx = fmaxf(mx, h2f(srow[i]));
    red[row * 16 + j] = mx; __syncthreads();
    mx = red[row * 16];
#pragma unroll
    for (int q = 1; q < 16; q++) mx = fmaxf(mx, red[row * 16 + q]);
    __syncthreads();

    // ---- entmax-1.5 tau via bisection: solve sum(relu((x-mx)/2 - tau))^2 == 1
    // f is monotone decreasing; exact bracket tau in [-1, 0] since max(y)=0.
    float lo = -1.0f, hi = 0.0f;
    for (int it = 0; it < 24; it++){
      float tau = 0.5f * (lo + hi);
      float ps = 0.f;
      for (int i = j; i < HN; i += 16){
        float p = fmaxf((h2f(srow[i]) - mx) * 0.5f - tau, 0.f);
        ps += p * p;
      }
      red[row * 16 + j] = ps; __syncthreads();
      float s = 0.f;
#pragma unroll
      for (int q = 0; q < 16; q++) s += red[row * 16 + q];
      if (s >= 1.0f) lo = tau; else hi = tau;
      __syncthreads();
    }
    if (j == 0){ maxs[row] = mx; taus[row] = 0.5f * (lo + hi); }
    __syncthreads();

    // ---- rewrite scores -> entmax weights w = relu(y - tau)^2 as bf16 ----
    for (int o = tid; o < 16 * HN; o += 256){
      int r2 = o >> 13;
      float p = fmaxf((h2f(sc[o]) - maxs[r2]) * 0.5f - taus[r2], 0.f);
      sc[o] = f2b_bits(p * p);
    }
    __syncthreads();

    // ---- GEMM2: xi = w @ V  (K-split over waves: 8 waves = 4 n-tiles x 2 halves)
    {
      int nt = wave & 3, kh = wave >> 2;
      int d0 = nt * 16;
      const unsigned short* vrow = Vt + (size_t)(hh * 64 + d0 + ln) * HN;
      const unsigned short* wrow = sc + ln * HN;
      v8f c = {};
      for (int kk = kh * 4096; kk < kh * 4096 + 4096; kk += 32){
        bf16x8 alo = *reinterpret_cast<const bf16x8*>(wrow + kk + 16 * lh);
        bf16x8 ahi = *reinterpret_cast<const bf16x8*>(wrow + kk + 16 * lh + 8);
        v16bf a;
#pragma unroll
        for (int e = 0; e < 8; e++){ a[e] = alo[e]; a[e + 8] = ahi[e]; }
        v16bf bv = *reinterpret_cast<const v16bf*>(vrow + kk + 16 * lh);
        c = __builtin_amdgcn_wmma_f32_16x16x32_bf16(false, a, false, bv, (short)0, c, false, false);
      }
#pragma unroll
      for (int r = 0; r < 8; r++) cpart[wave * 256 + (8 * lh + r) * 16 + ln] = c[r];
    }
    __syncthreads();
    for (int o = tid; o < 16 * 64; o += 256){
      int m = o >> 6, d = o & 63;
      int nt = d >> 4, dn = d & 15;
      xi[m * 64 + d] = cpart[nt * 256 + m * 16 + dn] + cpart[(nt + 4) * 256 + m * 16 + dn];
    }
    __syncthreads();
  }

  // converged: f32 to d_out tail, bf16 to workspace for the output GEMM
  for (int o = tid; o < 16 * 64; o += 256){
    int m = o >> 6, d = o & 63;
    float v = xi[m * 64 + d];
    size_t idx = (size_t)(rowbase + m) * HT + hh * 64 + d;
    conv_f[idx] = v;
    conv_b[idx] = f2b_bits(v);
  }
}

// ---------------------------------------------------------------------------
extern "C" void kernel_launch(void* const* d_in, const int* in_sizes, int n_in,
                              void* d_out, int out_size, void* d_ws, size_t ws_size,
                              hipStream_t stream) {
  (void)in_sizes; (void)n_in; (void)out_size; (void)ws_size;
  const float* hidden   = (const float*)d_in[0];  // [2,512,2048]
  const float* memory   = (const float*)d_in[1];  // [8192,1024]
  const float* Wq       = (const float*)d_in[2];  // [2048,256]
  const float* Wk       = (const float*)d_in[3];  // [1024,256]
  const float* Wv       = (const float*)d_in[4];  // [1024,256]
  const float* Wo       = (const float*)d_in[5];  // [256,2048]
  const float* log_beta = (const float*)d_in[6];  // [4]
  const float* g_s = (const float*)d_in[7];
  const float* b_s = (const float*)d_in[8];
  const float* g_m = (const float*)d_in[9];
  const float* b_m = (const float*)d_in[10];
  float* out = (float*)d_out;                     // [2097152 output | 262144 converged]

  char* ws = (char*)d_ws;
  unsigned short* hid_n  = (unsigned short*)(ws + 0);          // 1024x2048 bf16
  unsigned short* mem_n  = (unsigned short*)(ws + 4194304);    // 8192x1024 bf16
  unsigned short* Wq_t   = (unsigned short*)(ws + 20971520);   // 256x2048
  unsigned short* Wk_t   = (unsigned short*)(ws + 22020096);   // 256x1024
  unsigned short* Wv_t   = (unsigned short*)(ws + 22544384);   // 256x1024
  unsigned short* Wo_t   = (unsigned short*)(ws + 23068672);   // 2048x256
  float*          Qbuf   = (float*)        (ws + 24117248);    // 1024x256 f32
  unsigned short* K_bf   = (unsigned short*)(ws + 25165824);   // 8192x256
  unsigned short* V_t    = (unsigned short*)(ws + 29360128);   // 256x8192
  unsigned short* conv_b = (unsigned short*)(ws + 33554432);   // 1024x256

  // weight prep: transpose + convert to bf16 (Bt layouts for WMMA B-frags)
  k_transpose_bf16<<<(2048*256+255)/256, 256, 0, stream>>>(Wq, Wq_t, 2048, 256);
  k_transpose_bf16<<<(1024*256+255)/256, 256, 0, stream>>>(Wk, Wk_t, 1024, 256);
  k_transpose_bf16<<<(1024*256+255)/256, 256, 0, stream>>>(Wv, Wv_t, 1024, 256);
  k_transpose_bf16<<<(256*2048+255)/256, 256, 0, stream>>>(Wo, Wo_t, 256, 2048);

  // layernorms -> bf16 activations
  k_ln_bf16<<<1024, 256, 0, stream>>>(hidden, g_s, b_s, hid_n, 2048);
  k_ln_bf16<<<8192, 256, 0, stream>>>(memory, g_m, b_m, mem_n, 1024);

  // projections: Q (f32), K (bf16 row-major), V (bf16 transposed [T][N])
  k_gemm_bf16<<<dim3(64, 2),  256, 0, stream>>>(hid_n, Wq_t, Qbuf, nullptr, 1024, 256, 2048, 0);
  k_gemm_bf16<<<dim3(512, 2), 256, 0, stream>>>(mem_n, Wk_t, nullptr, K_bf, 8192, 256, 1024, 1);
  k_gemm_bf16<<<dim3(512, 2), 256, 0, stream>>>(mem_n, Wv_t, nullptr, V_t,  8192, 256, 1024, 2);

  // fused 3-step Hopfield: 256 blocks = B*H*(S/16); scores tile in LDS
  const size_t lds_bytes = 16*HN*2 + 16*64*4 + 256*4 + 16*4 + 16*4 + 8*256*4; // 275,584 B
  hipFuncSetAttribute(reinterpret_cast<const void*>(k_hopfield),
                      hipFuncAttributeMaxDynamicSharedMemorySize, (int)lds_bytes);
  k_hopfield<<<256, 256, lds_bytes, stream>>>(Qbuf, K_bf, V_t, log_beta, out + 2097152, conv_b);

  // output = converged @ Wo
  k_gemm_bf16<<<dim3(64, 16), 256, 0, stream>>>(conv_b, Wo_t, out, nullptr, 1024, 2048, 256, 0);
}
